// MultiHeadAttention_51900384805413
// MI455X (gfx1250) — matrix operand order repaired and checked by reading, not compiled, not run on hardware
//
#include <hip/hip_runtime.h>

#define BB 2
#define CC 2048
#define EE 1024
#define HH 16
#define DD 64
#define MMR (BB*CC)   /* 4096 rows */

typedef __attribute__((ext_vector_type(16))) __bf16 v16bf;
typedef __attribute__((ext_vector_type(8)))  __bf16 v8bf;
typedef __attribute__((ext_vector_type(8)))  float  v8f;
typedef int v4i __attribute__((vector_size(16)));   // matches builtin param type

// ---- gfx1250 data-mover feature detection (compile-safe) ----
#if defined(__has_builtin)
#if __has_builtin(__builtin_amdgcn_global_load_async_to_lds_b128)
#define USE_ASYNC 1
#endif
#if __has_builtin(__builtin_amdgcn_tensor_load_to_lds)
#define USE_TDM 1
#endif
#endif

static __device__ __forceinline__ __bf16 f2bf(float f) {
  unsigned u = __builtin_bit_cast(unsigned, f);
  u += 0x7FFFu + ((u >> 16) & 1u);          // round-to-nearest-even
  unsigned short s = (unsigned short)(u >> 16);
  return __builtin_bit_cast(__bf16, s);
}

static __device__ __forceinline__ v16bf frag16(const __bf16* p0, const __bf16* p1) {
  v8bf a = *(const v8bf*)p0;
  v8bf b = *(const v8bf*)p1;
  return __builtin_shufflevector(a, b, 0,1,2,3,4,5,6,7,8,9,10,11,12,13,14,15);
}

static __device__ __forceinline__ v8f wmma_bf16(v16bf a, v16bf b, v8f c) {
  return __builtin_amdgcn_wmma_f32_16x16x32_bf16(false, a, false, b, (short)0, c, false, false);
}

// 16-byte global -> LDS copy (async when available)
static __device__ __forceinline__ void cp16(const __bf16* g, __bf16* l) {
#ifdef USE_ASYNC
  __builtin_amdgcn_global_load_async_to_lds_b128(
      (__attribute__((address_space(1))) v4i*)g,
      (__attribute__((address_space(3))) v4i*)l, 0, 0);
#else
  *(v8bf*)l = *(const v8bf*)g;
#endif
}

static __device__ __forceinline__ void cp_wait() {
#ifdef USE_ASYNC
#if __has_builtin(__builtin_amdgcn_s_wait_asynccnt)
  __builtin_amdgcn_s_wait_asynccnt(0);
#else
  asm volatile("s_wait_asynccnt 0x0" ::: "memory");
#endif
#endif
}

#ifdef USE_TDM
typedef __attribute__((ext_vector_type(4))) unsigned tdm_g0_t;
typedef __attribute__((ext_vector_type(8))) int      tdm_g1_t;
typedef __attribute__((ext_vector_type(4))) int      tdm_g23_t;
typedef __attribute__((ext_vector_type(8))) int      tdm_g4_t;

static __device__ __forceinline__ unsigned lds_off(void* p) {
  return (unsigned)(unsigned long long)(__attribute__((address_space(3))) char*)p;
}

// 2-D tile load via Tensor Data Mover. Elements are 2 bytes (bf16).
// tile_x contiguous elements per row, tile_y rows, row stride in elements.
static __device__ __forceinline__ void tdm_load_2d(unsigned lds_byte_addr, const void* gptr,
                                                   unsigned tile_x, unsigned tile_y,
                                                   unsigned tensor_x, unsigned tensor_y,
                                                   unsigned row_stride_elems) {
  unsigned long long ga = (unsigned long long)(__SIZE_TYPE__)gptr;
  tdm_g0_t g0;
  g0[0] = 1u;                                                  // count=1, user mode
  g0[1] = lds_byte_addr;                                       // lds_addr
  g0[2] = (unsigned)(ga & 0xffffffffu);                        // global_addr[31:0]
  g0[3] = (unsigned)((ga >> 32) & 0x01ffffffu) | 0x80000000u;  // global_addr[56:32] | type=2
  tdm_g1_t g1;
  g1[0] = 0x00010000;                                          // data_size=1 (2B), mask=0
  g1[1] = (int)((tensor_x & 0xffffu) << 16);                   // tensor_dim0[15:0]
  g1[2] = (int)(((tensor_x >> 16) & 0xffffu) | ((tensor_y & 0xffffu) << 16));
  g1[3] = (int)(((tensor_y >> 16) & 0xffffu) | ((tile_x & 0xffffu) << 16));
  g1[4] = (int)(tile_y & 0xffffu);                             // tile_dim1 (tile_dim2=0)
  g1[5] = (int)row_stride_elems;                               // tensor_dim0_stride[31:0]
  g1[6] = 0;
  g1[7] = 0;
  tdm_g23_t z4 = {0, 0, 0, 0};
  tdm_g4_t  z8 = {0, 0, 0, 0, 0, 0, 0, 0};
  __builtin_amdgcn_tensor_load_to_lds(g0, g1, z4, z4, z8, 0);
}
#endif

// wait for K/V staging of the *current* block; if `more`, two descriptors
// (or 4 async ops) for the next block were just issued and may stay in flight.
static __device__ __forceinline__ void stage_wait(bool more) {
#ifdef USE_TDM
  if (more) __builtin_amdgcn_s_wait_tensorcnt(2);
  else      __builtin_amdgcn_s_wait_tensorcnt(0);
#elif defined(USE_ASYNC)
  if (more) __builtin_amdgcn_s_wait_asynccnt(4);
  else      __builtin_amdgcn_s_wait_asynccnt(0);
#else
  (void)more;
#endif
}

// ---------------- convert kernels ----------------

__global__ void cvt_x_kernel(const float* __restrict__ x, __bf16* __restrict__ xb, int n) {
  for (int i = blockIdx.x * blockDim.x + threadIdx.x; i < n; i += gridDim.x * blockDim.x)
    xb[i] = f2bf(x[i]);
}

// W [k][n] row-major fp32 -> Wt [n][k] bf16
__global__ void cvt_wt_kernel(const float* __restrict__ w, __bf16* __restrict__ wt) {
  int i = blockIdx.x * blockDim.x + threadIdx.x;   // exactly EE*EE threads
  int k = i >> 10, n = i & 1023;
  wt[n * EE + k] = f2bf(w[i]);
}

// ---------------- GEMM: out[m][n] = sum_k A[m][k] * Bt[n][k] ----------------
// Double-buffered 128x128x32 tile, 8 waves (2M x 4N), wave tile 64x32.
// mode 0: bf16 row-major out; mode 1: bf16 V^T-per-head out; mode 2: f32 out + bias.

__global__ __launch_bounds__(256)
void gemm_bf16_kernel(const __bf16* __restrict__ A, const __bf16* __restrict__ Bt,
                      void* __restrict__ outp, const float* __restrict__ bias, int mode) {
  const int K = EE, N = EE;
  __shared__ __align__(16) __bf16 As[2][128][32];
  __shared__ __align__(16) __bf16 Bs[2][128][32];

  const int bm = blockIdx.x * 128;
  const int bn = blockIdx.y * 128;
  const int t = threadIdx.x;
  const int wave = t >> 5, lane = t & 31;
  const int ln = lane & 15, hi = lane >> 4;
  const int wm = (wave & 1) * 64;
  const int wn = (wave >> 1) * 32;

  v8f acc[4][2] = {};

  const int lr = t >> 1, lc = (t & 1) * 16;                 // 32B per thread per tile
  const __bf16* gA = A  + (size_t)(bm + lr) * K + lc;
  const __bf16* gB = Bt + (size_t)(bn + lr) * K + lc;

  // preload buffer 0
  cp16(gA, &As[0][lr][lc]); cp16(gA + 8, &As[0][lr][lc + 8]);
  cp16(gB, &Bs[0][lr][lc]); cp16(gB + 8, &Bs[0][lr][lc + 8]);
  cp_wait();
  __syncthreads();

  int cur = 0;
  for (int kb = 0; kb < K; kb += 32) {
    if (kb + 32 < K) {      // issue next tile into the other buffer
      gA += 32; gB += 32;
      cp16(gA, &As[cur ^ 1][lr][lc]); cp16(gA + 8, &As[cur ^ 1][lr][lc + 8]);
      cp16(gB, &Bs[cur ^ 1][lr][lc]); cp16(gB + 8, &Bs[cur ^ 1][lr][lc + 8]);
      __builtin_prefetch(gA + 32);
      __builtin_prefetch(gB + 32);
    }

    v16bf af[4], bfv[2];
#pragma unroll
    for (int i = 0; i < 4; ++i)
      af[i] = frag16(&As[cur][wm + 16*i + ln][16*hi], &As[cur][wm + 16*i + ln][16*hi + 8]);
#pragma unroll
    for (int j = 0; j < 2; ++j)
      bfv[j] = frag16(&Bs[cur][wn + 16*j + ln][16*hi], &Bs[cur][wn + 16*j + ln][16*hi + 8]);
#pragma unroll
    for (int i = 0; i < 4; ++i)
#pragma unroll
      for (int j = 0; j < 2; ++j)
        acc[i][j] = wmma_bf16(af[i], bfv[j], acc[i][j]);

    cp_wait();
    __syncthreads();
    cur ^= 1;
  }

#pragma unroll
  for (int i = 0; i < 4; ++i)
#pragma unroll
    for (int j = 0; j < 2; ++j)
#pragma unroll
      for (int r = 0; r < 8; ++r) {
        int m = bm + wm + 16*i + r + 8*hi;
        int n = bn + wn + 16*j + ln;
        float v = acc[i][j][r];
        if (mode == 0) {
          ((__bf16*)outp)[(size_t)m * N + n] = f2bf(v);
        } else if (mode == 1) {
          int b = m >> 11, c = m & 2047;
          int h = n >> 6,  d = n & 63;
          ((__bf16*)outp)[(((size_t)(b * HH + h) * DD + d) << 11) + c] = f2bf(v);
        } else {
          ((float*)outp)[(size_t)m * N + n] = v + bias[n];
        }
      }
}

// ---------------- flash attention ----------------
// grid: (C/128, H, B). 256 threads = 8 waves; wave w owns query rows qb+16w..+15.
// K/V^T staging is double-buffered: TDM (or async) loads for block i+1 overlap
// the WMMA/softmax work on block i (in-order completion => s_wait_tensorcnt 2).

__global__ __launch_bounds__(256)
void attn_kernel(const __bf16* __restrict__ Qb, const __bf16* __restrict__ Kb,
                 const __bf16* __restrict__ Vtb, __bf16* __restrict__ Ctx) {
  __shared__ __align__(16) __bf16 Qs[128][64];      // 16KB
  __shared__ __align__(16) __bf16 Ks[2][64][64];    // 2x8KB  [key][d]
  __shared__ __align__(16) __bf16 Vs[2][64][64];    // 2x8KB  [d][key]
  __shared__ __align__(16) __bf16 Ps[8][16][64];    // 16KB per-wave P tiles

  const int qb = blockIdx.x * 128;
  const int h  = blockIdx.y;
  const int b  = blockIdx.z;
  const int t = threadIdx.x;
  const int wave = t >> 5, lane = t & 31;
  const int ln = lane & 15, hi = lane >> 4;

  const __bf16* Qbase  = Qb  + (size_t)(b * CC) * EE + h * DD;          // row stride EE
  const __bf16* Kbase  = Kb  + (size_t)(b * CC) * EE + h * DD;
  const __bf16* Vtbase = Vtb + (size_t)(b * HH + h) * DD * CC;          // [d][c]

#ifdef USE_TDM
#define STAGE_KV(buf, kb_) do {                                                             \
    if (t < 32) {                                                                           \
      tdm_load_2d(lds_off(&Ks[buf][0][0]), Kbase + (size_t)(kb_) * EE,                      \
                  64, 64, 64, (unsigned)(CC - (kb_)), EE);                                  \
      tdm_load_2d(lds_off(&Vs[buf][0][0]), Vtbase + (kb_),                                  \
                  64, 64, (unsigned)(CC - (kb_)), DD, CC);                                  \
    } } while (0)
#else
#define STAGE_KV(buf, kb_) do {                                                             \
      int r_ = t >> 2, c0_ = (t & 3) * 16;                                                  \
      const __bf16* gk_ = Kbase + (size_t)((kb_) + r_) * EE + c0_;                          \
      cp16(gk_,     &Ks[buf][r_][c0_]);                                                     \
      cp16(gk_ + 8, &Ks[buf][r_][c0_ + 8]);                                                 \
      const __bf16* gv_ = Vtbase + (size_t)r_ * CC + (kb_) + c0_;                           \
      cp16(gv_,     &Vs[buf][r_][c0_]);                                                     \
      cp16(gv_ + 8, &Vs[buf][r_][c0_ + 8]);                                                 \
    } while (0)
#endif

  // ---- stage Q block: 128 rows x 64 d ----
#ifdef USE_TDM
  if (t < 32) {
    tdm_load_2d(lds_off(&Qs[0][0]), Qbase + (size_t)qb * EE,
                /*tile*/ 64, 128, /*tensor*/ 64, (unsigned)(CC - qb), EE);
    __builtin_amdgcn_s_wait_tensorcnt(0);
  }
#else
  {
    int r = t >> 1, c0 = (t & 1) * 32;
    const __bf16* g = Qbase + (size_t)(qb + r) * EE + c0;
    cp16(g,      &Qs[r][c0]);
    cp16(g + 8,  &Qs[r][c0 + 8]);
    cp16(g + 16, &Qs[r][c0 + 16]);
    cp16(g + 24, &Qs[r][c0 + 24]);
  }
  cp_wait();
#endif
  __syncthreads();

  v16bf aQ[2];
#pragma unroll
  for (int s = 0; s < 2; ++s)
    aQ[s] = frag16(&Qs[wave*16 + ln][32*s + 16*hi], &Qs[wave*16 + ln][32*s + 16*hi + 8]);

  v8f ctx[4] = {};
  float rmax[8], rsum[8];
#pragma unroll
  for (int r = 0; r < 8; ++r) { rmax[r] = -1e30f; rsum[r] = 0.f; }

  const int qrow0 = qb + wave * 16 + 8 * hi;   // + r per register
  const int nkb = (qb + 128) >> 6;             // causal frontier in 64-key blocks

  STAGE_KV(0, 0);                              // prologue: block 0 into buffer 0
  int cur = 0;
  for (int ib = 0; ib < nkb; ++ib) {
    const int kb = ib << 6;
    if (ib + 1 < nkb) {                        // overlap next block's DMA with compute
      STAGE_KV(cur ^ 1, kb + 64);
      stage_wait(true);
    } else {
      stage_wait(false);
    }
    __syncthreads();

    // S tiles: j = key sub-tile of 16
    v8f S[4];
    float nmax[8];
#pragma unroll
    for (int r = 0; r < 8; ++r) nmax[r] = rmax[r];

#pragma unroll
    for (int j = 0; j < 4; ++j) {
      v16bf bk0 = frag16(&Ks[cur][16*j + ln][16*hi],      &Ks[cur][16*j + ln][16*hi + 8]);
      v16bf bk1 = frag16(&Ks[cur][16*j + ln][32 + 16*hi], &Ks[cur][16*j + ln][40 + 16*hi]);
      v8f s_ = {};
      s_ = wmma_bf16(aQ[0], bk0, s_);
      s_ = wmma_bf16(aQ[1], bk1, s_);
      int key = kb + 16*j + ln;
#pragma unroll
      for (int r = 0; r < 8; ++r) {
        float v = s_[r] * 0.03125f;               // 1/sqrt(E) = 1/32
        if (key > qrow0 + r) v = -1e30f;          // causal mask
        s_[r] = v;
        nmax[r] = fmaxf(nmax[r], v);
      }
      S[j] = s_;
    }

    // row-max reduction across the 16 lanes of each half
#pragma unroll
    for (int r = 0; r < 8; ++r) {
#pragma unroll
      for (int off = 1; off < 16; off <<= 1)
        nmax[r] = fmaxf(nmax[r], __shfl_xor(nmax[r], off, 16));
    }

    float fac[8];
#pragma unroll
    for (int r = 0; r < 8; ++r) {
      fac[r] = __expf(rmax[r] - nmax[r]);
      rmax[r] = nmax[r];
      rsum[r] *= fac[r];
    }
#pragma unroll
    for (int td = 0; td < 4; ++td)
#pragma unroll
      for (int r = 0; r < 8; ++r) ctx[td][r] *= fac[r];

    // P = exp(S - max); lane-local partial row sums; D-layout -> A-layout via LDS
#pragma unroll
    for (int j = 0; j < 4; ++j)
#pragma unroll
      for (int r = 0; r < 8; ++r) {
        float p = __expf(S[j][r] - rmax[r]);
        rsum[r] += p;
        Ps[wave][r + 8*hi][16*j + ln] = f2bf(p);
      }

    v16bf aP0 = frag16(&Ps[wave][ln][16*hi],      &Ps[wave][ln][16*hi + 8]);
    v16bf aP1 = frag16(&Ps[wave][ln][32 + 16*hi], &Ps[wave][ln][40 + 16*hi]);

#pragma unroll
    for (int td = 0; td < 4; ++td) {
      v16bf bv0 = frag16(&Vs[cur][16*td + ln][16*hi],      &Vs[cur][16*td + ln][16*hi + 8]);
      v16bf bv1 = frag16(&Vs[cur][16*td + ln][32 + 16*hi], &Vs[cur][16*td + ln][40 + 16*hi]);
      ctx[td] = wmma_bf16(aP0, bv0, ctx[td]);
      ctx[td] = wmma_bf16(aP1, bv1, ctx[td]);
    }
    __syncthreads();   // all LDS reads done before this buffer is restaged
    cur ^= 1;
  }

  // finalize: reduce lane-local row sums, normalize, store ctx (bf16)
  float inv[8];
#pragma unroll
  for (int r = 0; r < 8; ++r) {
    float s = rsum[r];
#pragma unroll
    for (int off = 1; off < 16; off <<= 1) s += __shfl_xor(s, off, 16);
    inv[r] = 1.0f / s;
  }
#pragma unroll
  for (int td = 0; td < 4; ++td)
#pragma unroll
    for (int r = 0; r < 8; ++r) {
      int c = qb + wave*16 + r + 8*hi;
      int e = h * DD + 16*td + ln;
      Ctx[(size_t)(b * CC + c) * EE + e] = f2bf(ctx[td][r] * inv[r]);
    }
#undef STAGE_KV
}

// ---------------- host launcher ----------------

extern "C" void kernel_launch(void* const* d_in, const int* in_sizes, int n_in,
                              void* d_out, int out_size, void* d_ws, size_t ws_size,
                              hipStream_t stream) {
  (void)in_sizes; (void)n_in; (void)out_size; (void)ws_size;
  const float* x  = (const float*)d_in[0];
  const float* Wq = (const float*)d_in[1];
  const float* Wk = (const float*)d_in[2];
  const float* Wv = (const float*)d_in[3];
  const float* Wo = (const float*)d_in[4];
  const float* bo = (const float*)d_in[5];

  char* w = (char*)d_ws;
  __bf16* Xb  = (__bf16*)(w);                       // 4M elems, 8MB
  __bf16* Wqt = (__bf16*)(w + (size_t)(8u << 20));  // 1M elems each, 2MB each
  __bf16* Wkt = Wqt + (1u << 20);
  __bf16* Wvt = Wkt + (1u << 20);
  __bf16* Wot = Wvt + (1u << 20);
  __bf16* Qb  = (__bf16*)(w + (size_t)(16u << 20)); // 4M elems, 8MB each
  __bf16* Kb  = Qb + (1u << 22);
  __bf16* Vtb = Kb + (1u << 22);
  __bf16* Ctx = Vtb + (1u << 22);

  cvt_x_kernel<<<4096, 256, 0, stream>>>(x, Xb, MMR * EE);
  cvt_wt_kernel<<<4096, 256, 0, stream>>>(Wq, Wqt);
  cvt_wt_kernel<<<4096, 256, 0, stream>>>(Wk, Wkt);
  cvt_wt_kernel<<<4096, 256, 0, stream>>>(Wv, Wvt);
  cvt_wt_kernel<<<4096, 256, 0, stream>>>(Wo, Wot);

  dim3 gg(MMR / 128, EE / 128);
  gemm_bf16_kernel<<<gg, 256, 0, stream>>>(Xb, Wqt, (void*)Qb,  nullptr, 0);
  gemm_bf16_kernel<<<gg, 256, 0, stream>>>(Xb, Wkt, (void*)Kb,  nullptr, 0);
  gemm_bf16_kernel<<<gg, 256, 0, stream>>>(Xb, Wvt, (void*)Vtb, nullptr, 1);

  attn_kernel<<<dim3(CC / 128, HH, BB), 256, 0, stream>>>(Qb, Kb, Vtb, Ctx);

  gemm_bf16_kernel<<<gg, 256, 0, stream>>>(Ctx, Wot, d_out, bo, 2);
}
